// MultiheadAttentionWithRPE_38594576122474
// MI455X (gfx1250) — compile-verified, matrix operand order checked by reading, not run on hardware
//
#include <hip/hip_runtime.h>
#include <hip/hip_bf16.h>

// ---------------------------------------------------------------------------
// MultiheadAttentionWithRPE for MI455X (gfx1250), wave32 + WMMA f16->f32.
// B=2, N=2048, D=1024, H=16, HD=64, S=2048.
// d_out = [ out (B*N*D f32) | attn_weights (B*H*N*N f32) ]
// ---------------------------------------------------------------------------

#define BSZ   2
#define NSEQ  2048
#define DMOD  1024
#define NH    16
#define HDIM  64
#define NTOK  (BSZ * NSEQ)      // 4096
#define D3    (3 * DMOD)        // 3072

typedef __attribute__((ext_vector_type(16))) _Float16 v16h;
typedef __attribute__((ext_vector_type(8)))  float    v8f;
typedef __attribute__((ext_vector_type(2)))  _Float16 h2f;

// ---- WMMA fragment loaders -------------------------------------------------
// A (16x32 f16) / B (32x16 f16, sourced from a row-major matrix whose rows
// are the WMMA columns): lane&15 selects the row, the 16 half-slots hold K
// pairs per ISA 7.12.2: K = 2v + 8*((v>=4) + (lane>=16)).

__device__ __forceinline__ v16h frag_from_f16(const _Float16* __restrict__ p,
                                              int ld, int row0, int k0) {
    const int lane = threadIdx.x & 31;
    const _Float16* base =
        p + (size_t)(row0 + (lane & 15)) * ld + k0 + ((lane >> 4) ? 8 : 0);
    v16h a;
#pragma unroll
    for (int v = 0; v < 8; ++v) {
        const int k = 2 * v + ((v >= 4) ? 8 : 0);
        const h2f t = *(const h2f*)(base + k);
        a[2 * v]     = t.x;
        a[2 * v + 1] = t.y;
    }
    return a;
}

__device__ __forceinline__ v16h frag_from_f32(const float* __restrict__ p,
                                              int ld, int row0, int k0) {
    const int lane = threadIdx.x & 31;
    const float* base =
        p + (size_t)(row0 + (lane & 15)) * ld + k0 + ((lane >> 4) ? 8 : 0);
    v16h a;
#pragma unroll
    for (int v = 0; v < 8; ++v) {
        const int k = 2 * v + ((v >= 4) ? 8 : 0);
        const float2 t = *(const float2*)(base + k);
        a[2 * v]     = (_Float16)t.x;
        a[2 * v + 1] = (_Float16)t.y;
    }
    return a;
}

// ---------------------------------------------------------------------------
// Kernel 1: fused QKV projection. qkv = query @ W^T + b.
// Each wave: 16 rows x 64 cols (1 A-tile x 4 B-tiles, 4 accumulators).
// Q gets the 1/sqrt(HD) scale folded in. V stored transposed per (b,h).
// ---------------------------------------------------------------------------
__global__ void k_qkv_proj(const float* __restrict__ query,   // [B,N,D]
                           const float* __restrict__ W,       // [3D,D]
                           const float* __restrict__ bias,    // [3D]
                           _Float16* __restrict__ Qws,        // [B,H,N,HD]
                           _Float16* __restrict__ Kws,        // [B,H,N,HD]
                           _Float16* __restrict__ Vt) {       // [B,H,HD,N]
    const int wave = blockIdx.x * (blockDim.x >> 5) + (threadIdx.x >> 5);
    const int lane = threadIdx.x & 31;
    const int njg  = D3 / 64;           // 48 column groups of 64
    const int mt   = wave / njg;
    const int jg   = wave % njg;
    const int m0   = mt * 16;
    const int j0   = jg * 64;

    v8f c[4] = {};
    for (int k0 = 0; k0 < DMOD; k0 += 32) {
        v16h a = frag_from_f32(query, DMOD, m0, k0);
        if (k0 + 32 < DMOD) {
            __builtin_prefetch(query + (size_t)(m0 + (lane & 15)) * DMOD + k0 + 32, 0, 1);
            __builtin_prefetch(W     + (size_t)(j0 + (lane & 15)) * DMOD + k0 + 32, 0, 1);
        }
#pragma unroll
        for (int js = 0; js < 4; ++js) {
            v16h b = frag_from_f32(W, DMOD, j0 + js * 16, k0);
            c[js] = __builtin_amdgcn_wmma_f32_16x16x32_f16(false, a, false, b,
                                                           (short)0, c[js],
                                                           false, false);
        }
    }

#pragma unroll
    for (int js = 0; js < 4; ++js) {
        const int   j     = j0 + js * 16 + (lane & 15);
        const float bj    = bias[j];
        const int   which = j / DMOD;       // 0=Q 1=K 2=V (uniform per subtile)
        const int   jj    = j % DMOD;
        const int   hh    = jj / HDIM;      // uniform per subtile
        const int   hd    = jj % HDIM;
#pragma unroll
        for (int r = 0; r < 8; ++r) {
            const int   m   = m0 + r + 8 * (lane >> 4);
            const int   bb  = m / NSEQ;
            const int   n   = m % NSEQ;
            const float val = c[js][r] + bj;
            if (which == 0) {
                Qws[(((size_t)(bb * NH + hh)) * NSEQ + n) * HDIM + hd] =
                    (_Float16)(val * 0.125f);   // HD^-0.5 = 1/8
            } else if (which == 1) {
                Kws[(((size_t)(bb * NH + hh)) * NSEQ + n) * HDIM + hd] =
                    (_Float16)val;
            } else {
                Vt[(((size_t)(bb * NH + hh)) * HDIM + hd) * NSEQ + n] =
                    (_Float16)val;
            }
        }
    }
}

// ---------------------------------------------------------------------------
// Kernel 2: raw scores = Q K^T + rpe_bias  -> attn region of d_out (f32).
// Each wave: 1 q-tile x 4 consecutive k-tiles; Q fragments loaded once.
// ---------------------------------------------------------------------------
__global__ void k_scores(const _Float16* __restrict__ Qws,
                         const _Float16* __restrict__ Kws,
                         const int*   __restrict__ relidx,    // [N,N]
                         const float* __restrict__ rpe,       // [S,H]
                         float* __restrict__ attn) {          // [B,H,N,N]
    const int wave = blockIdx.x * (blockDim.x >> 5) + (threadIdx.x >> 5);
    const int lane = threadIdx.x & 31;
    const int nkg  = NSEQ / 64;                    // 32 k-groups of 64
    const int tph  = (NSEQ / 16) * nkg;            // 4096 wave-jobs / (b,h)
    const int bh   = wave / tph;
    const int t    = wave % tph;
    const int qt   = t / nkg;
    const int kg   = t % nkg;
    const int h    = bh % NH;

    const _Float16* Qp = Qws + (size_t)bh * NSEQ * HDIM;
    const _Float16* Kp = Kws + (size_t)bh * NSEQ * HDIM;

    const v16h a0 = frag_from_f16(Qp, HDIM, qt * 16, 0);
    const v16h a1 = frag_from_f16(Qp, HDIM, qt * 16, 32);

    v8f c[4] = {};
#pragma unroll
    for (int js = 0; js < 4; ++js) {
        const int krow = kg * 64 + js * 16;
        v16h b0 = frag_from_f16(Kp, HDIM, krow, 0);
        v16h b1 = frag_from_f16(Kp, HDIM, krow, 32);
        c[js] = __builtin_amdgcn_wmma_f32_16x16x32_f16(false, a0, false, b0,
                                                       (short)0, c[js], false, false);
        c[js] = __builtin_amdgcn_wmma_f32_16x16x32_f16(false, a1, false, b1,
                                                       (short)0, c[js], false, false);
    }

    float* out = attn + (size_t)bh * NSEQ * NSEQ;
#pragma unroll
    for (int js = 0; js < 4; ++js) {
        const int kk = kg * 64 + js * 16 + (lane & 15);
#pragma unroll
        for (int r = 0; r < 8; ++r) {
            const int   qi = qt * 16 + r + 8 * (lane >> 4);
            const int   ix = relidx[(size_t)qi * NSEQ + kk];
            const float bv = rpe[(size_t)ix * NH + h];
            out[(size_t)qi * NSEQ + kk] = c[js][r] + bv;
        }
    }
}

// ---------------------------------------------------------------------------
// Kernel 3: row softmax in place over the last axis (len 2048).
// One 256-thread block per row; 8 elements per thread.
// ---------------------------------------------------------------------------
__global__ void k_softmax(float* __restrict__ attn) {
    float* p = attn + (size_t)blockIdx.x * NSEQ;
    __shared__ float red[256];
    const int tid = threadIdx.x;

    float vals[8];
    float m = -1e30f;
#pragma unroll
    for (int i = 0; i < 8; ++i) {
        vals[i] = p[tid + i * 256];
        m = fmaxf(m, vals[i]);
    }
    red[tid] = m;
    __syncthreads();
    for (int s = 128; s > 0; s >>= 1) {
        if (tid < s) red[tid] = fmaxf(red[tid], red[tid + s]);
        __syncthreads();
    }
    m = red[0];
    __syncthreads();

    float sum = 0.f;
#pragma unroll
    for (int i = 0; i < 8; ++i) {
        vals[i] = __expf(vals[i] - m);
        sum += vals[i];
    }
    red[tid] = sum;
    __syncthreads();
    for (int s = 128; s > 0; s >>= 1) {
        if (tid < s) red[tid] += red[tid + s];
        __syncthreads();
    }
    const float inv = 1.0f / red[0];
#pragma unroll
    for (int i = 0; i < 8; ++i) p[tid + i * 256] = vals[i] * inv;
}

// ---------------------------------------------------------------------------
// Kernel 4: attn_out = P @ V  (A = normalized attn f32->f16, B = V^T in LDS).
// One block = (b,h, ct, q-group of 8 tiles). The 16x2048 V^T strip (64 KB) is
// staged into LDS once per block with GLOBAL_LOAD_ASYNC_TO_LDS_B128 and
// reused by all 8 q-tiles. Each wave shares its B fragment across 2 q-tiles.
// ---------------------------------------------------------------------------
#define PV_QTPB 8   // q-tiles per block (2 per wave)

__global__ void k_pv(const float* __restrict__ attn,        // [B,H,N,N] normalized
                     const _Float16* __restrict__ Vt,       // [B,H,HD,N]
                     _Float16* __restrict__ aout) {         // [B,N,D]
    const int lane = threadIdx.x & 31;
    const int wib  = threadIdx.x >> 5;                 // wave in block, 0..3
    const int nQG  = (NSEQ / 16) / PV_QTPB;            // 16 q-groups
    const int qg   = blockIdx.x % nQG;
    const int ct   = (blockIdx.x / nQG) % (HDIM / 16); // hd tile
    const int bh   = blockIdx.x / (nQG * (HDIM / 16));
    const int h    = bh % NH;
    const int bb   = bh / NH;

    __shared__ _Float16 vs[16 * NSEQ];                 // 64 KB V^T strip

    // --- async stage V^T rows [ct*16, ct*16+16) x [0, NSEQ) into LDS ------
    const _Float16* Vsrc = Vt + ((size_t)bh * HDIM + ct * 16) * NSEQ;
    for (int i = threadIdx.x; i < (16 * NSEQ) / 8; i += blockDim.x) {
        const _Float16* src = Vsrc + (size_t)i * 8;           // 16B chunk
        const unsigned  dst =
            (unsigned)(unsigned long long)(const void*)&vs[i * 8];
        asm volatile("global_load_async_to_lds_b128 %0, %1, off"
                     :: "v"(dst), "v"(src) : "memory");
    }
    asm volatile("s_wait_asynccnt 0x0" ::: "memory");
    __syncthreads();

    const float* P   = attn + (size_t)bh * NSEQ * NSEQ;
    const int    qt0 = qg * PV_QTPB + wib * 2;
    const int    qt1 = qt0 + 1;

    v8f c0 = {}, c1 = {};
    for (int k0 = 0; k0 < NSEQ; k0 += 32) {
        v16h b  = frag_from_f16(vs, NSEQ, 0, k0);      // ds_load from LDS
        v16h a0 = frag_from_f32(P, NSEQ, qt0 * 16, k0);
        v16h a1 = frag_from_f32(P, NSEQ, qt1 * 16, k0);
        c0 = __builtin_amdgcn_wmma_f32_16x16x32_f16(false, a0, false, b,
                                                    (short)0, c0, false, false);
        c1 = __builtin_amdgcn_wmma_f32_16x16x32_f16(false, a1, false, b,
                                                    (short)0, c1, false, false);
    }

    const int hd = ct * 16 + (lane & 15);
#pragma unroll
    for (int r = 0; r < 8; ++r) {
        const int qi0 = qt0 * 16 + r + 8 * (lane >> 4);
        const int qi1 = qt1 * 16 + r + 8 * (lane >> 4);
        aout[((size_t)(bb * NSEQ + qi0)) * DMOD + h * HDIM + hd] = (_Float16)c0[r];
        aout[((size_t)(bb * NSEQ + qi1)) * DMOD + h * HDIM + hd] = (_Float16)c1[r];
    }
}

// ---------------------------------------------------------------------------
// Kernel 5: out = attn_out @ Wout^T + bias  -> d_out[0 : B*N*D] (f32).
// Each wave: 16 rows x 64 cols (1 A-tile x 4 B-tiles).
// ---------------------------------------------------------------------------
__global__ void k_outproj(const _Float16* __restrict__ aout,  // [B*N, D] f16
                          const float* __restrict__ Wout,     // [D, D]
                          const float* __restrict__ bias,     // [D]
                          float* __restrict__ out) {          // [B*N, D]
    const int wave = blockIdx.x * (blockDim.x >> 5) + (threadIdx.x >> 5);
    const int lane = threadIdx.x & 31;
    const int njg  = DMOD / 64;   // 16 column groups
    const int mt   = wave / njg;
    const int jg   = wave % njg;
    const int j0   = jg * 64;

    v8f c[4] = {};
    for (int k0 = 0; k0 < DMOD; k0 += 32) {
        v16h a = frag_from_f16(aout, DMOD, mt * 16, k0);
#pragma unroll
        for (int js = 0; js < 4; ++js) {
            v16h b = frag_from_f32(Wout, DMOD, j0 + js * 16, k0);
            c[js] = __builtin_amdgcn_wmma_f32_16x16x32_f16(false, a, false, b,
                                                           (short)0, c[js],
                                                           false, false);
        }
    }

#pragma unroll
    for (int js = 0; js < 4; ++js) {
        const int   j  = j0 + js * 16 + (lane & 15);
        const float bj = bias[j];
#pragma unroll
        for (int r = 0; r < 8; ++r) {
            const int m = mt * 16 + r + 8 * (lane >> 4);
            out[(size_t)m * DMOD + j] = c[js][r] + bj;
        }
    }
}

// ---------------------------------------------------------------------------
extern "C" void kernel_launch(void* const* d_in, const int* in_sizes, int n_in,
                              void* d_out, int out_size, void* d_ws, size_t ws_size,
                              hipStream_t stream) {
    (void)in_sizes; (void)n_in; (void)out_size; (void)ws_size;

    const float* query  = (const float*)d_in[0];
    // d_in[1] (key), d_in[2] (value) are unused by the reference (QKV all
    // come from `query` through the fused in-projection).
    const int*   relidx = (const int*)  d_in[3];
    const float* Wqkv   = (const float*)d_in[4];
    const float* bqkv   = (const float*)d_in[5];
    const float* Wout   = (const float*)d_in[6];
    const float* bout   = (const float*)d_in[7];
    const float* rpe    = (const float*)d_in[8];

    float* out  = (float*)d_out;
    float* attn = out + (size_t)BSZ * NSEQ * DMOD;   // [B,H,N,N]

    const size_t qkvElems = (size_t)BSZ * NH * NSEQ * HDIM;   // 4.19M f16 each
    _Float16* Qws  = (_Float16*)d_ws;
    _Float16* Kws  = Qws + qkvElems;
    _Float16* Vt   = Kws + qkvElems;
    _Float16* aout = Vt  + qkvElems;                           // [B,N,D] f16

    const int threads = 128;   // 4 waves per block

    // 1) QKV projection: 256 m-tiles x 48 j-groups = 12288 waves / 4
    k_qkv_proj<<<12288 / 4, threads, 0, stream>>>(query, Wqkv, bqkv, Qws, Kws, Vt);

    // 2) Scores + RPE bias: B*H * 128 q-tiles * 32 k-groups = 131072 waves / 4
    k_scores<<<131072 / 4, threads, 0, stream>>>(Qws, Kws, relidx, rpe, attn);

    // 3) Softmax in place: one block per row, B*H*N rows
    k_softmax<<<BSZ * NH * NSEQ, 256, 0, stream>>>(attn);

    // 4) P @ V: B*H * 4 hd-tiles * 16 q-groups = 2048 blocks
    k_pv<<<BSZ * NH * (HDIM / 16) * ((NSEQ / 16) / PV_QTPB), threads, 0, stream>>>(
        attn, Vt, aout);

    // 5) Output projection: 256 m-tiles x 16 j-groups = 4096 waves / 4
    k_outproj<<<4096 / 4, threads, 0, stream>>>(aout, Wout, bout, out);
}
